// FastWeightsHomeostaticReentryLayer_47708496724669
// MI455X (gfx1250) — matrix operand order repaired and checked by reading, not compiled, not run on hardware
//
#include <hip/hip_runtime.h>

// ---------------- problem constants ----------------
#define BB 8
#define SS 2048
#define DD 1024
#define MM (BB * SS)          // 16384 rows of Y / out
#define RFLOAT 32.0f
#define ALPHA 0.2f
#define BETA 0.1f
#define GAMMA 0.1f

// scan segmentation: (1-ALPHA)^128 = 0.8^128 ~ 4e-13 << f32 eps
#define SEG 128
#define WARM 128
#define NSEG (SS / SEG)       // 16

// GEMM tiling
#define BM 128
#define BN 128
#define KT 64
#define LDST 72               // bf16 elems per LDS row: 144 bytes (TDM pads 64+8)
#define LDSE (BM * LDST)      // elements per tile buffer (9216)

typedef __attribute__((ext_vector_type(16))) __bf16 v16bf;
typedef __attribute__((ext_vector_type(8)))  float v8f;
typedef unsigned int u32x4 __attribute__((ext_vector_type(4)));
typedef int          i32x4 __attribute__((ext_vector_type(4)));
typedef int          i32x8 __attribute__((ext_vector_type(8)));

#if __has_builtin(__builtin_amdgcn_tensor_load_to_lds) && \
    __has_builtin(__builtin_amdgcn_s_wait_tensorcnt)
#define USE_TDM 1
#else
#define USE_TDM 0
#endif

// ---------------- helpers ----------------
__device__ __forceinline__ unsigned short f2bf(float f) {
    union { float f; unsigned int u; } c; c.f = f;
    unsigned int u = c.u + 0x7FFFu + ((c.u >> 16) & 1u);   // round-to-nearest-even
    return (unsigned short)(u >> 16);
}

// reduce two floats across a 256-thread block (wave32 shuffle tree + LDS)
__device__ __forceinline__ void block_reduce2(float& a, float& b, float* red) {
    #pragma unroll
    for (int off = 16; off > 0; off >>= 1) {
        a += __shfl_xor(a, off, 32);
        b += __shfl_xor(b, off, 32);
    }
    const int w = threadIdx.x >> 5, l = threadIdx.x & 31;
    if (l == 0) { red[w * 2] = a; red[w * 2 + 1] = b; }
    __syncthreads();
    float ra = 0.f, rb = 0.f;
    #pragma unroll
    for (int i = 0; i < 8; ++i) { ra += red[i * 2]; rb += red[i * 2 + 1]; }
    a = ra; b = rb;
    __syncthreads();
}

#if USE_TDM
// Issue one TDM 2D tile DMA: rows x KT bf16 from row-major global (stride DD)
// into LDS at lds_byte_addr, HW-padded to LDST bf16 per row.
__device__ __forceinline__ void tdm_load_tile(const unsigned short* gsrc,
                                              unsigned lds_byte_addr,
                                              int tensor_rows) {
    const unsigned long long ga = (unsigned long long)(size_t)gsrc;
    u32x4 g0 = { 1u,                                  // count=1, user descriptor
                 lds_byte_addr,                       // D#.lds_addr
                 (unsigned)ga,                        // global_addr[31:0]
                 (unsigned)((ga >> 32) & 0x01FFFFFFu) | (2u << 30) }; // [56:32] | type=2
    i32x8 g1 = { (1 << 16) | (1 << 20) | (4 << 22) | (3 << 25),
                 //  data_size=2B | pad_en | pad_interval=32dw(128B) | pad_amount=4dw(16B)
                 (int)((DD & 0xFFFF) << 16),          // tensor_dim0[15:0]
                 (int)((DD >> 16) | ((tensor_rows & 0xFFFF) << 16)), // td0 hi | td1 lo
                 (int)((tensor_rows >> 16) | (KT << 16)),            // td1 hi | tile_dim0
                 BM,                                  // tile_dim1=128, tile_dim2=0
                 DD,                                  // tensor_dim0_stride lo
                 0, 0 };
    i32x4 gz4 = { 0, 0, 0, 0 };
#if defined(__clang_major__) && (__clang_major__ >= 23)
    i32x8 gz8 = { 0, 0, 0, 0, 0, 0, 0, 0 };
    __builtin_amdgcn_tensor_load_to_lds(g0, g1, gz4, gz4, gz8, 0);
#else
    __builtin_amdgcn_tensor_load_to_lds(g0, g1, gz4, gz4, 0);
#endif
}
#endif

// ---------------- kernel 1: segmented scan -> Y (bf16) ----------------
__global__ __launch_bounds__(256)
void fw_scan_kernel(const float* __restrict__ q, const float* __restrict__ k,
                    const float* __restrict__ vv, unsigned short* __restrict__ Ybf) {
    __shared__ float red[16];
    const int b   = blockIdx.x / NSEG;
    const int seg = blockIdx.x % NSEG;
    const int t_out0 = seg * SEG;
    const int t0     = (t_out0 >= WARM) ? (t_out0 - WARM) : 0;
    const int t_end  = t_out0 + SEG;
    const int tid = threadIdx.x;

    float u[4] = {0.f, 0.f, 0.f, 0.f};
    float v[4] = {0.f, 0.f, 0.f, 0.f};

    for (int t = t0; t < t_end; ++t) {
        const size_t base = ((size_t)b * SS + t) * DD;
        float qv[4], kv[4], wv[4];
        float sq = 0.f, sk = 0.f;
        #pragma unroll
        for (int i = 0; i < 4; ++i) {
            const int d = tid + i * 256;
            qv[i] = q[base + d];
            kv[i] = k[base + d];
            wv[i] = vv[base + d];
            sq += qv[i] * qv[i];
            sk += kv[i] * kv[i];
        }
        block_reduce2(sq, sk, red);
        const float rq = 1.0f / (sqrtf(sq) + 1e-6f);
        const float rk = 1.0f / (sqrtf(sk) + 1e-6f);

        float sdot = 0.f, su2 = 0.f;
        #pragma unroll
        for (int i = 0; i < 4; ++i) {
            u[i] = (1.0f - ALPHA) * u[i] + ALPHA * qv[i] * rq;
            v[i] = (1.0f - ALPHA) * v[i] + ALPHA * kv[i] * rk;
            sdot += v[i] * wv[i];
            su2  += u[i] * u[i];
        }
        block_reduce2(sdot, su2, red);

        if (t >= t_out0) {
            const float c0   = RFLOAT * sdot;                 // y = c0 * u
            const float norm = fabsf(c0) * sqrtf(su2);
            const float n    = fmaxf(norm, 1e-6f);
            const float c    = c0 / (1.0f + BETA * (n - 1.0f));
            #pragma unroll
            for (int i = 0; i < 4; ++i) {
                const int d = tid + i * 256;
                Ybf[base + d] = f2bf(c * u[i]);
            }
        }
    }
}

// ---------------- kernel 2: W (f32) -> bf16 ----------------
__global__ __launch_bounds__(256)
void fw_cvtW_kernel(const float* __restrict__ W, unsigned short* __restrict__ Wbf) {
    const size_t i = ((size_t)blockIdx.x * 256 + threadIdx.x) * 4;
    const float4 w = *(const float4*)(W + i);
    ushort4 o;
    o.x = f2bf(w.x); o.y = f2bf(w.y); o.z = f2bf(w.z); o.w = f2bf(w.w);
    *(ushort4*)(Wbf + i) = o;
}

// ---------------- kernel 3: out = x + GAMMA * Y @ W^T  (bf16 WMMA + TDM) ----------------
// out[m,n] = x[m,n] + GAMMA * sum_k Y[m,k] * W[n,k]
// Double-buffered K-tiles staged into LDS by the Tensor Data Mover.
__global__ __launch_bounds__(256)
void fw_gemm_kernel(const unsigned short* __restrict__ Ybf,
                    const unsigned short* __restrict__ Wbf,
                    const float* __restrict__ x, float* __restrict__ out) {
    __shared__ unsigned short smem[4 * LDSE];   // [buf][A|B][BM*LDST]

    const int m0 = blockIdx.y * BM;
    const int n0 = blockIdx.x * BN;
    const int tid  = threadIdx.x;
    const int lane = tid & 31;
    const int wave = tid >> 5;       // 0..7
    const int wm = wave >> 1;        // 0..3 -> M offset wm*32
    const int wn = wave & 1;         // 0..1 -> N offset wn*64
    const int lrow  = lane & 15;
    const int lhalf = lane >> 4;

    v8f acc[2][4] = {};

#if USE_TDM
    const unsigned lds_base = (unsigned)(size_t)&smem[0];   // low 32 bits = LDS offset
    // wave0 DMAs the A tile, wave1 DMAs the B tile
    #define ISSUE_TILES(buf_, k0_)                                              \
        do {                                                                    \
            if (wave == 0)                                                      \
                tdm_load_tile(Ybf + (size_t)m0 * DD + (k0_),                    \
                              lds_base + (unsigned)(((buf_)*2 + 0) * LDSE * 2), \
                              MM);                                              \
            else if (wave == 1)                                                 \
                tdm_load_tile(Wbf + (size_t)n0 * DD + (k0_),                    \
                              lds_base + (unsigned)(((buf_)*2 + 1) * LDSE * 2), \
                              DD);                                              \
        } while (0)

    ISSUE_TILES(0, 0);
#endif

    for (int k0 = 0; k0 < DD; k0 += KT) {
        const int buf = (k0 / KT) & 1;
#if USE_TDM
        if (k0 + KT < DD) {
            ISSUE_TILES(buf ^ 1, k0 + KT);                 // prefetch next K-tile
            __builtin_amdgcn_s_wait_tensorcnt(1);          // current tile landed
        } else {
            __builtin_amdgcn_s_wait_tensorcnt(0);
        }
        __syncthreads();                                   // publish LDS to all waves
        const unsigned short* At = smem + (buf * 2 + 0) * LDSE;
        const unsigned short* Bt = smem + (buf * 2 + 1) * LDSE;
#else
        __syncthreads();
        const unsigned short* At = smem;
        const unsigned short* Bt = smem + LDSE;
        #pragma unroll
        for (int j = 0; j < 4; ++j) {
            const int chunk = tid + j * 256;
            const int row = chunk >> 3;
            const int kc  = (chunk & 7) * 8;
            *(uint4*)(&smem[row * LDST + kc]) =
                *(const uint4*)(Ybf + (size_t)(m0 + row) * DD + k0 + kc);
            *(uint4*)(&smem[LDSE + row * LDST + kc]) =
                *(const uint4*)(Wbf + (size_t)(n0 + row) * DD + k0 + kc);
        }
        __syncthreads();
        (void)buf;
#endif

        #pragma unroll
        for (int kk = 0; kk < KT; kk += 32) {
            // A fragment (16x32 bf16): lane m=lrow; K chunks at kk+lhalf*8 and +16
            v16bf afrag[2];
            #pragma unroll
            for (int tm = 0; tm < 2; ++tm) {
                const int row = wm * 32 + tm * 16 + lrow;
                const int cb  = kk + lhalf * 8;
                union { uint4 u[2]; v16bf v; } tmp;
                tmp.u[0] = *(const uint4*)(&At[row * LDST + cb]);
                tmp.u[1] = *(const uint4*)(&At[row * LDST + cb + 16]);
                afrag[tm] = tmp.v;
            }
            // B fragment (32x16 bf16): lane n=lrow; K chunks at kk+lhalf*8 and +16
            v16bf bfrag[4];
            #pragma unroll
            for (int tn = 0; tn < 4; ++tn) {
                const int col = wn * 64 + tn * 16 + lrow;
                const int cb  = kk + lhalf * 8;
                union { uint4 u[2]; v16bf v; } tmp;
                tmp.u[0] = *(const uint4*)(&Bt[col * LDST + cb]);
                tmp.u[1] = *(const uint4*)(&Bt[col * LDST + cb + 16]);
                bfrag[tn] = tmp.v;
            }
            #pragma unroll
            for (int tm = 0; tm < 2; ++tm)
                #pragma unroll
                for (int tn = 0; tn < 4; ++tn)
                    acc[tm][tn] = __builtin_amdgcn_wmma_f32_16x16x32_bf16(
                        false, afrag[tm], false, bfrag[tn],
                        (short)0, acc[tm][tn], false, false);
        }
        __syncthreads();   // all waves done reading this buffer before reuse
    }

    // epilogue: C layout — VGPR j holds M=j (lanes 0-15) / M=j+8 (lanes 16-31), N=lane&15
    #pragma unroll
    for (int tm = 0; tm < 2; ++tm)
        #pragma unroll
        for (int tn = 0; tn < 4; ++tn)
            #pragma unroll
            for (int j = 0; j < 8; ++j) {
                const int gm = m0 + wm * 32 + tm * 16 + j + lhalf * 8;
                const int gn = n0 + wn * 64 + tn * 16 + lrow;
                const size_t idx = (size_t)gm * DD + gn;
                out[idx] = x[idx] + GAMMA * acc[tm][tn][j];
            }
}

// ---------------- launch ----------------
extern "C" void kernel_launch(void* const* d_in, const int* in_sizes, int n_in,
                              void* d_out, int out_size, void* d_ws, size_t ws_size,
                              hipStream_t stream) {
    const float* x  = (const float*)d_in[0];
    const float* q  = (const float*)d_in[1];
    const float* k  = (const float*)d_in[2];
    const float* vv = (const float*)d_in[3];
    const float* W  = (const float*)d_in[4];
    float* out = (float*)d_out;

    unsigned short* Ybf = (unsigned short*)d_ws;                                 // 32 MB
    unsigned short* Wbf = (unsigned short*)((char*)d_ws + (size_t)MM * DD * 2);  // +2 MB

    fw_cvtW_kernel<<<dim3((DD * DD) / (256 * 4)), dim3(256), 0, stream>>>(W, Wbf);
    fw_scan_kernel<<<dim3(BB * NSEG), dim3(256), 0, stream>>>(q, k, vv, Ybf);
    fw_gemm_kernel<<<dim3(DD / BN, MM / BM), dim3(256), 0, stream>>>(Ybf, Wbf, x, out);
}
